// GNN_36318243455158
// MI455X (gfx1250) — matrix operand order repaired and checked by reading, not compiled, not run on hardware
//
#include <hip/hip_runtime.h>
#include <hip/hip_bf16.h>

typedef __attribute__((ext_vector_type(16))) __bf16 v16bf;
typedef __attribute__((ext_vector_type(8)))  __bf16 v8bf;
typedef __attribute__((ext_vector_type(8)))  float  v8f;

#define NN 100000
#define NE 1600000
#define BN_EPS 1e-5f

// ---------------- histogram of dst (int atomics) ----------------
__global__ void hist_kernel(const int* __restrict__ dst, int* __restrict__ cnt) {
    int e = blockIdx.x * blockDim.x + threadIdx.x;
    if (e < NE) atomicAdd(&cnt[dst[e]], 1);
}

// ---------------- single-block exclusive scan over NN counts ----------------
__global__ void scan_kernel(const int* __restrict__ cnt, int* __restrict__ offs,
                            int* __restrict__ cursor) {
    __shared__ int sdata[256];
    __shared__ int sbase;
    int tid = threadIdx.x;
    if (tid == 0) sbase = 0;
    __syncthreads();
    for (int base = 0; base < NN; base += 256) {
        int i = base + tid;
        int v = (i < NN) ? cnt[i] : 0;
        int x = v;
        sdata[tid] = x;
        __syncthreads();
        for (int d = 1; d < 256; d <<= 1) {
            int t = (tid >= d) ? sdata[tid - d] : 0;
            __syncthreads();
            x += t;
            sdata[tid] = x;
            __syncthreads();
        }
        if (i < NN) {
            int o = sbase + (x - v);
            offs[i] = o;
            cursor[i] = o;
        }
        __syncthreads();
        if (tid == 255) sbase += sdata[255];
        __syncthreads();
    }
}

// ---------------- bucket edges by destination (counting sort) ----------------
__global__ void bucket_kernel(const int* __restrict__ src, const int* __restrict__ dst,
                              int* __restrict__ cursor, int* __restrict__ srcSorted) {
    int e = blockIdx.x * blockDim.x + threadIdx.x;
    if (e >= NE) return;
    int d = dst[e];
    int pos = atomicAdd(&cursor[d], 1);
    srcSorted[pos] = src[e];
}

// ---------------- wave-per-node gather-aggregate -> bf16 xcat agg half ----------------
// Lane owns one float4 (4 channels) of the 128-float row; registers accumulate over the
// node's in-edges (no atomics), then mean is written as packed bf16x4 (8B store).
__global__ void aggregate_kernel(const float* __restrict__ feat, const int* __restrict__ srcSorted,
                                 const int* __restrict__ offs, const int* __restrict__ cnt,
                                 __bf16* __restrict__ xcat) {
    int node = blockIdx.x * (blockDim.x >> 5) + (threadIdx.x >> 5);
    if (node >= NN) return;
    int lane = threadIdx.x & 31;
    int beg = offs[node];
    int n   = cnt[node];

    float4 a0 = {0.f, 0.f, 0.f, 0.f}, a1 = {0.f, 0.f, 0.f, 0.f};
    int j = 0;
    for (; j + 1 < n; j += 2) {
        int s0 = srcSorted[beg + j];
        int s1 = srcSorted[beg + j + 1];
        float4 v0 = ((const float4*)(feat + (size_t)s0 * 128))[lane];
        float4 v1 = ((const float4*)(feat + (size_t)s1 * 128))[lane];
        a0.x += v0.x; a0.y += v0.y; a0.z += v0.z; a0.w += v0.w;
        a1.x += v1.x; a1.y += v1.y; a1.z += v1.z; a1.w += v1.w;
    }
    if (j < n) {
        int s0 = srcSorted[beg + j];
        float4 v0 = ((const float4*)(feat + (size_t)s0 * 128))[lane];
        a0.x += v0.x; a0.y += v0.y; a0.z += v0.z; a0.w += v0.w;
    }
    float inv = 1.0f / fmaxf((float)n, 1.0f);
    union { __bf16 b[4]; uint2 u; } pk;
    pk.b[0] = (__bf16)((a0.x + a1.x) * inv);
    pk.b[1] = (__bf16)((a0.y + a1.y) * inv);
    pk.b[2] = (__bf16)((a0.z + a1.z) * inv);
    pk.b[3] = (__bf16)((a0.w + a1.w) * inv);
    *(uint2*)(xcat + (size_t)node * 256 + 128 + lane * 4) = pk.u;
}

// ---------------- one-time: xcat self half <- bf16(x) (layer 0 only) ----------------
__global__ void cvt_self_kernel(const float* __restrict__ feat, __bf16* __restrict__ xcat) {
    long idx = (long)blockIdx.x * blockDim.x + threadIdx.x;   // over NN*128
    if (idx >= (long)NN * 128) return;
    int node = (int)(idx >> 7);
    int c    = (int)(idx & 127);
    xcat[(size_t)node * 256 + c] = (__bf16)feat[idx];
}

// ---------------- pack [w_self; w_neigh] (256 x Ncols) into WMMA B-fragment order ----------------
// frag(kc, t): lane L, elem i  ->  K = kc*32 + (L/16)*8 + (i<8 ? i : 16 + i - 8),  Nc = t*16 + (L%16)
__global__ void pack_w_kernel(const float* __restrict__ w_self, const float* __restrict__ w_neigh,
                              __bf16* __restrict__ wpack, int Ncols) {
    int NT = Ncols >> 4;
    int total = 8 * NT * 512;
    int idx = blockIdx.x * blockDim.x + threadIdx.x;
    if (idx >= total) return;
    int i    = idx & 15;
    int lane = (idx >> 4) & 31;
    int frag = idx >> 9;
    int t  = frag % NT;
    int kc = frag / NT;
    int k = kc * 32 + (lane >> 4) * 8 + (i < 8 ? i : 16 + (i - 8));
    int n = t * 16 + (lane & 15);
    float v = (k < 128) ? w_self[k * Ncols + n] : w_neigh[(k - 128) * Ncols + n];
    wpack[idx] = (__bf16)v;
}

// ---------------- bf16 WMMA GEMM: out[NN, NT*16] = xcat[NN,256] @ wcat + bias ----------------
// Weights staged in LDS once per block (8 waves); B frags via ds_load (DScnt), A via
// global_load_b128 (LOADcnt) with explicit K double-buffer -> WMMAs overlap both streams.
template <int NT>
__global__ __launch_bounds__(256) void gemm_wmma_kernel(const __bf16* __restrict__ xcat,
                                                        const __bf16* __restrict__ wpack,
                                                        const float* __restrict__ bias,
                                                        float* __restrict__ out) {
    const int Ncols = NT * 16;
    __shared__ __attribute__((aligned(32))) __bf16 wlds[8 * NT * 512];   // 64KB / 32KB

    {
        const uint4* gsrc = (const uint4*)wpack;
        uint4* ldst = (uint4*)wlds;
        const int nvec = NT * 512;
        for (int i = threadIdx.x; i < nvec; i += 256) ldst[i] = gsrc[i];
    }
    __syncthreads();

    int wave = threadIdx.x >> 5;
    int tile = blockIdx.x * 8 + wave;                 // wave-uniform guard -> EXEC all ones
    if (tile >= NN / 16) return;
    int lane = threadIdx.x & 31;
    int m    = lane & 15;
    int half = lane >> 4;
    const __bf16* arow = xcat + (size_t)(tile * 16 + m) * 256;

    v8f acc[NT];
#pragma unroll
    for (int t = 0; t < NT; ++t) acc[t] = (v8f){};

    // A-frag per lane (16-bit A 16x32 layout): elems 0..7 = K half*8..+7, elems 8..15 = +16.
    v8bf lo = *(const v8bf*)(arow + half * 8);
    v8bf hi = *(const v8bf*)(arow + half * 8 + 16);

#pragma unroll
    for (int kc = 0; kc < 8; ++kc) {
        v16bf bfr[NT];
#pragma unroll
        for (int t = 0; t < NT; ++t)
            bfr[t] = *(const v16bf*)(&wlds[(((unsigned)kc * NT + t) * 32 + lane) * 16]);

        v8bf nlo = lo, nhi = hi;
        if (kc < 7) {
            int kb = (kc + 1) * 32 + half * 8;
            nlo = *(const v8bf*)(arow + kb);
            nhi = *(const v8bf*)(arow + kb + 16);
        }

        v16bf a;
#pragma unroll
        for (int i = 0; i < 8; ++i) { a[i] = lo[i]; a[i + 8] = hi[i]; }
#pragma unroll
        for (int t = 0; t < NT; ++t)
            acc[t] = __builtin_amdgcn_wmma_f32_16x16x32_bf16(
                false, a, false, bfr[t], (short)0, acc[t], false, false);

        lo = nlo; hi = nhi;
    }

    // epilogue: C/D layout -> VGPR r holds (M = r + 8*half, N = lane%16); fuse bias.
#pragma unroll
    for (int t = 0; t < NT; ++t) {
        int n = t * 16 + m;
        float bv = bias[n];
#pragma unroll
        for (int r = 0; r < 8; ++r) {
            int row = tile * 16 + r + 8 * half;
            out[(size_t)row * Ncols + n] = acc[t][r] + bv;
        }
    }
}

// ---------------- BN stats: per-channel sum / sumsq (coalesced strips + atomics) ----------------
__global__ void bn_stats_kernel(const float* __restrict__ h, float* __restrict__ stats) {
    int c = threadIdx.x;   // blockDim.x == 128 channels
    float s = 0.f, s2 = 0.f;
    for (int node = blockIdx.x; node < NN; node += gridDim.x) {
        float v = h[(size_t)node * 128 + c];
        s += v;
        s2 += v * v;
    }
    atomicAdd(&stats[c], s);
    atomicAdd(&stats[128 + c], s2);
}

// ---------------- fused BN + ReLU; writes f32 feat and bf16 self-half of next xcat ----------------
__global__ void bn_relu_kernel(const float* __restrict__ h, const float* __restrict__ stats,
                               const float* __restrict__ gamma, const float* __restrict__ beta,
                               float* __restrict__ outf, __bf16* __restrict__ xcat_self) {
    long idx = (long)blockIdx.x * blockDim.x + threadIdx.x;
    if (idx >= (long)NN * 128) return;
    int node = (int)(idx >> 7);
    int c    = (int)(idx & 127);
    float mean = stats[c] * (1.0f / NN);
    float var  = stats[128 + c] * (1.0f / NN) - mean * mean;
    float inv  = rsqrtf(var + BN_EPS);
    float v = (h[idx] - mean) * inv * gamma[c] + beta[c];
    v = v > 0.f ? v : 0.f;
    outf[idx] = v;
    xcat_self[(size_t)node * 256 + c] = (__bf16)v;
}

// ---------------- wave-per-node log_softmax over 64 channels ----------------
__global__ void logsoftmax_kernel(const float* __restrict__ h, float* __restrict__ out) {
    int node = blockIdx.x * (blockDim.x >> 5) + (threadIdx.x >> 5);
    if (node >= NN) return;
    int lane = threadIdx.x & 31;
    float a = h[(size_t)node * 64 + lane];
    float b = h[(size_t)node * 64 + 32 + lane];
    float mx = fmaxf(a, b);
#pragma unroll
    for (int off = 16; off >= 1; off >>= 1) mx = fmaxf(mx, __shfl_xor(mx, off, 32));
    float s = __expf(a - mx) + __expf(b - mx);
#pragma unroll
    for (int off = 16; off >= 1; off >>= 1) s += __shfl_xor(s, off, 32);
    float lse = mx + __logf(s);
    out[(size_t)node * 64 + lane]      = a - lse;
    out[(size_t)node * 64 + 32 + lane] = b - lse;
}

extern "C" void kernel_launch(void* const* d_in, const int* in_sizes, int n_in,
                              void* d_out, int out_size, void* d_ws, size_t ws_size,
                              hipStream_t stream) {
    const float* x    = (const float*)d_in[0];
    const int*   eidx = (const int*)d_in[1];
    const int*   src  = eidx;
    const int*   dst  = eidx + NE;
    const float* w0s = (const float*)d_in[2];
    const float* w0n = (const float*)d_in[3];
    const float* b0  = (const float*)d_in[4];
    const float* g0  = (const float*)d_in[5];
    const float* be0 = (const float*)d_in[6];
    const float* w1s = (const float*)d_in[7];
    const float* w1n = (const float*)d_in[8];
    const float* b1  = (const float*)d_in[9];
    const float* g1  = (const float*)d_in[10];
    const float* be1 = (const float*)d_in[11];
    const float* w2s = (const float*)d_in[12];
    const float* w2n = (const float*)d_in[13];
    const float* b2  = (const float*)d_in[14];
    float* out = (float*)d_out;

    char* ws = (char*)d_ws;
    size_t off = 0;
    auto alloc = [&](size_t bytes) -> void* {
        void* p = ws + off;
        off += (bytes + 255) & ~(size_t)255;
        return p;
    };
    float*  featB     = (float*)alloc((size_t)NN * 128 * 4);   // 51.2 MB
    float*  hA        = (float*)alloc((size_t)NN * 128 * 4);   // 51.2 MB
    __bf16* xcat      = (__bf16*)alloc((size_t)NN * 256 * 2);  // 51.2 MB
    int*    cnt       = (int*)alloc((size_t)NN * 4);
    int*    offs      = (int*)alloc((size_t)NN * 4);
    int*    cursor    = (int*)alloc((size_t)NN * 4);
    int*    srcSorted = (int*)alloc((size_t)NE * 4);           // 6.4 MB
    __bf16* wp0       = (__bf16*)alloc(8 * 8 * 512 * 2);
    __bf16* wp1       = (__bf16*)alloc(8 * 8 * 512 * 2);
    __bf16* wp2       = (__bf16*)alloc(8 * 4 * 512 * 2);
    float*  stats     = (float*)alloc(256 * 4);

    const int elemBlocks = (int)(((long)NN * 128 + 255) / 256);
    const int gemmBlocks = (NN / 16 + 7) / 8;   // 8 waves/block, wave per 16-row tile
    const int nodeBlocks = (NN + 7) / 8;        // 8 waves/block, wave per node

    // ---- one-time per launch: CSR-ize the graph by destination ----
    hipMemsetAsync(cnt, 0, (size_t)NN * 4, stream);
    hist_kernel<<<(NE + 255) / 256, 256, 0, stream>>>(dst, cnt);
    scan_kernel<<<1, 256, 0, stream>>>(cnt, offs, cursor);
    bucket_kernel<<<(NE + 255) / 256, 256, 0, stream>>>(src, dst, cursor, srcSorted);

    // weight packing + layer-0 self-half convert
    pack_w_kernel<<<(8 * 8 * 512 + 255) / 256, 256, 0, stream>>>(w0s, w0n, wp0, 128);
    pack_w_kernel<<<(8 * 8 * 512 + 255) / 256, 256, 0, stream>>>(w1s, w1n, wp1, 128);
    pack_w_kernel<<<(8 * 4 * 512 + 255) / 256, 256, 0, stream>>>(w2s, w2n, wp2, 64);
    cvt_self_kernel<<<elemBlocks, 256, 0, stream>>>(x, xcat);

    // -------- layer 0 --------
    aggregate_kernel<<<nodeBlocks, 256, 0, stream>>>(x, srcSorted, offs, cnt, xcat);
    gemm_wmma_kernel<8><<<gemmBlocks, 256, 0, stream>>>(xcat, wp0, b0, hA);
    hipMemsetAsync(stats, 0, 256 * 4, stream);
    bn_stats_kernel<<<512, 128, 0, stream>>>(hA, stats);
    bn_relu_kernel<<<elemBlocks, 256, 0, stream>>>(hA, stats, g0, be0, featB, xcat);

    // -------- layer 1 --------
    aggregate_kernel<<<nodeBlocks, 256, 0, stream>>>(featB, srcSorted, offs, cnt, xcat);
    gemm_wmma_kernel<8><<<gemmBlocks, 256, 0, stream>>>(xcat, wp1, b1, hA);
    hipMemsetAsync(stats, 0, 256 * 4, stream);
    bn_stats_kernel<<<512, 128, 0, stream>>>(hA, stats);
    bn_relu_kernel<<<elemBlocks, 256, 0, stream>>>(hA, stats, g1, be1, featB, xcat);

    // -------- layer 2 + log_softmax --------
    aggregate_kernel<<<nodeBlocks, 256, 0, stream>>>(featB, srcSorted, offs, cnt, xcat);
    gemm_wmma_kernel<4><<<gemmBlocks, 256, 0, stream>>>(xcat, wp2, b2, hA);
    logsoftmax_kernel<<<nodeBlocks, 256, 0, stream>>>(hA, out);
}
